// VRWKV_SpatialMix_50345606644109
// MI455X (gfx1250) — matrix operand order repaired and checked by reading, not compiled, not run on hardware
//
#include <hip/hip_runtime.h>
#include <hip/hip_bf16.h>

// ---------------------------------------------------------------------------
// Problem constants (match reference: B=8, T=4096, C=768)
// ---------------------------------------------------------------------------
#define BB   8
#define TT   4096
#define CC   768
#define MM   (BB * TT)          // 32768 rows
#define NCAT (3 * CC)           // 2304 concatenated k|v|r columns
#define CHL  64                 // chunk length for WKV scan
#define NCH  (TT / CHL)         // 64 chunks

// GEMM tiling: block 128M x 128N x 32K, 8 waves each owning 32M x 64N
#define BM 128
#define BN 128
#define BK 32
#define NKIT (CC / BK)          // 24 K-steps
#define LDA 40                  // LDS row pitch in ushorts (80B: 16B-aligned, low bank conflict)

#if __has_builtin(__builtin_amdgcn_global_load_async_to_lds_b128) && \
    __has_builtin(__builtin_amdgcn_s_wait_asynccnt)
#define HAVE_ASYNC_LDS 1
#else
#define HAVE_ASYNC_LDS 0
#endif

typedef __attribute__((ext_vector_type(16))) __bf16 v16bf;
typedef __attribute__((ext_vector_type(8)))  float  v8f;
typedef __attribute__((ext_vector_type(4)))  int    v4i;

union FragBF { v16bf v; unsigned int u[8]; };

// float -> bf16 bits, round-to-nearest-even
__device__ __forceinline__ unsigned short f2bf(float f) {
  unsigned int u = __builtin_bit_cast(unsigned int, f);
  u += 0x7FFFu + ((u >> 16) & 1u);
  return (unsigned short)(u >> 16);
}
__device__ __forceinline__ float bf2f(unsigned short h) {
  unsigned int u = ((unsigned int)h) << 16;
  return __builtin_bit_cast(float, u);
}
__device__ __forceinline__ void cvt_store8(unsigned short* dst, const float4 f) {
  unsigned int lo = ((unsigned int)f2bf(f.y) << 16) | f2bf(f.x);
  unsigned int hi = ((unsigned int)f2bf(f.w) << 16) | f2bf(f.z);
  unsigned int* d = reinterpret_cast<unsigned int*>(dst);
  d[0] = lo; d[1] = hi;
}

#if HAVE_ASYNC_LDS
__device__ __forceinline__ void async_copy16(const unsigned short* g, unsigned short* l) {
  __builtin_amdgcn_global_load_async_to_lds_b128(
      (__attribute__((address_space(1))) v4i*)(g),
      (__attribute__((address_space(3))) v4i*)(l), 0, 0);
}
#endif

// ---------------------------------------------------------------------------
// WMMA fragment gathers from LDS, per CDNA5 ISA 7.12.2 bf16 layouts (wave32)
// ---------------------------------------------------------------------------
// A (16x32, MxK): lane L: M=L&15, hi=L>>4. VGPR j holds K = (j<4?0:16)+hi*8+2*(j&3) (+1)
__device__ __forceinline__ v16bf load_a_frag(const unsigned short* As, int mbase, int lane) {
  const int m  = mbase + (lane & 15);
  const int hi = lane >> 4;
  const unsigned int* row = reinterpret_cast<const unsigned int*>(As + m * LDA);
  FragBF r;
#pragma unroll
  for (int j = 0; j < 8; ++j) {
    const int kb = ((j & 4) << 2) + hi * 8 + ((j & 3) << 1);
    r.u[j] = row[kb >> 1];
  }
  return r.v;
}
// B (32x16, KxN), stored in LDS as Bs[n][k] (row n of W): lane L: N=L&15, hi=L>>4.
// VGPR j holds K = (j<4?0:16)+hi*8+2*(j&3) (+1)
__device__ __forceinline__ v16bf load_b_frag(const unsigned short* Bs, int nbase, int lane) {
  const int n  = nbase + (lane & 15);
  const int hi = lane >> 4;
  const unsigned int* row = reinterpret_cast<const unsigned int*>(Bs + n * LDA);
  FragBF r;
#pragma unroll
  for (int j = 0; j < 8; ++j) r.u[j] = row[((j & 4) << 1) + hi * 4 + (j & 3)];
  return r.v;
}

// One 32K step for a 32M x 64N wave tile: 8 WMMAs
__device__ __forceinline__ void mma_step(const unsigned short* As, const unsigned short* Bs,
                                         int wm, int wn, int lane, v8f (&acc)[2][4]) {
  v16bf a[2], b[4];
  a[0] = load_a_frag(As, wm * 32,      lane);
  a[1] = load_a_frag(As, wm * 32 + 16, lane);
#pragma unroll
  for (int j = 0; j < 4; ++j) b[j] = load_b_frag(Bs, wn * 64 + j * 16, lane);
#pragma unroll
  for (int i = 0; i < 2; ++i)
#pragma unroll
    for (int j = 0; j < 4; ++j)
      acc[i][j] = __builtin_amdgcn_wmma_f32_16x16x32_bf16(false, a[i], false, b[j],
                                                          (short)0, acc[i][j], false, false);
}

// ---------------------------------------------------------------------------
// GEMM #1: D[m, n] = sum_k x[m,k] * Wcat[n,k];  n in [0,2304)
// Epilogue: n<768 -> k_f32 ; n<1536 -> v_f32 ; else sr=sigmoid -> bf16
// grid = (NCAT/BN, MM/BM) = (18, 256), block = 256
// ---------------------------------------------------------------------------
__global__ __launch_bounds__(256)
void vrwkv_gemm_kvr(const float* __restrict__ X,
                    const float* __restrict__ Wk,
                    const float* __restrict__ Wv,
                    const float* __restrict__ Wr,
                    float* __restrict__ kOut,
                    float* __restrict__ vOut,
                    unsigned short* __restrict__ srOut) {
  __shared__ unsigned short As[2][BM * LDA];
  __shared__ unsigned short Bs[2][BN * LDA];

  const int tid  = threadIdx.x;
  const int lane = tid & 31;
  const int wid  = tid >> 5;
  const int wm   = wid & 3;   // 4 waves over M (32 rows each)
  const int wn   = wid >> 2;  // 2 waves over N (64 cols each)

  const int n0 = blockIdx.x * BN;
  const int m0 = blockIdx.y * BM;

  const int wsel  = n0 / CC;
  const int nloc0 = n0 - wsel * CC;
  const float* W = (wsel == 0) ? Wk : ((wsel == 1) ? Wv : Wr);

  const int r8 = tid >> 3;            // 0..31
  const int c4 = (tid & 7) << 2;      // 0..28 step 4

  v8f acc[2][4] = {};
  float4 aReg[4], bReg[4];

  // prologue: tile 0
#pragma unroll
  for (int s = 0; s < 4; ++s) {
    aReg[s] = *reinterpret_cast<const float4*>(X + (size_t)(m0 + r8 + s * 32) * CC + c4);
    bReg[s] = *reinterpret_cast<const float4*>(W + (size_t)(nloc0 + r8 + s * 32) * CC + c4);
  }
#pragma unroll
  for (int s = 0; s < 4; ++s) {
    cvt_store8(&As[0][(r8 + s * 32) * LDA + c4], aReg[s]);
    cvt_store8(&Bs[0][(r8 + s * 32) * LDA + c4], bReg[s]);
  }
  __syncthreads();

  for (int it = 0; it < NKIT; ++it) {
    const int cur = it & 1;
    const int kt  = (it + 1) * BK;
    if (it + 1 < NKIT) {
#pragma unroll
      for (int s = 0; s < 4; ++s) {
        aReg[s] = *reinterpret_cast<const float4*>(X + (size_t)(m0 + r8 + s * 32) * CC + kt + c4);
        bReg[s] = *reinterpret_cast<const float4*>(W + (size_t)(nloc0 + r8 + s * 32) * CC + kt + c4);
      }
    }
    mma_step(As[cur], Bs[cur], wm, wn, lane, acc);
    if (it + 1 < NKIT) {
#pragma unroll
      for (int s = 0; s < 4; ++s) {
        cvt_store8(&As[cur ^ 1][(r8 + s * 32) * LDA + c4], aReg[s]);
        cvt_store8(&Bs[cur ^ 1][(r8 + s * 32) * LDA + c4], bReg[s]);
      }
    }
    __syncthreads();
  }

  // Epilogue: D layout: lane L: N=L&15, M = vgpr + 8*(L>>4)
  const int lane15 = lane & 15;
  const int hi8 = (lane >> 4) * 8;
#pragma unroll
  for (int i = 0; i < 2; ++i) {
#pragma unroll
    for (int j = 0; j < 4; ++j) {
      const int nloc = nloc0 + wn * 64 + j * 16 + lane15;
      const size_t mb = (size_t)m0 + wm * 32 + i * 16 + hi8;
      const v8f a = acc[i][j];
#pragma unroll
      for (int r = 0; r < 8; ++r) {
        const size_t off = (mb + r) * CC + nloc;
        const float val = a[r];
        if (wsel == 0)      kOut[off] = val;
        else if (wsel == 1) vOut[off] = val;
        else                srOut[off] = f2bf(1.0f / (1.0f + __expf(-val)));
      }
    }
  }
}

// ---------------------------------------------------------------------------
// GEMM #2: out[m,n] = sum_k z[m,k](bf16) * Wo[n,k];  N = 768, fp32 out
// A tile copied global->LDS with CDNA5 async-to-LDS when available.
// grid = (768/BN, MM/BM) = (6, 256)
// ---------------------------------------------------------------------------
__global__ __launch_bounds__(256)
void vrwkv_gemm_out(const unsigned short* __restrict__ Z,
                    const float* __restrict__ Wo,
                    float* __restrict__ Out) {
  __shared__ unsigned short As[2][BM * LDA];
  __shared__ unsigned short Bs[2][BN * LDA];

  const int tid  = threadIdx.x;
  const int lane = tid & 31;
  const int wid  = tid >> 5;
  const int wm   = wid & 3;
  const int wn   = wid >> 2;

  const int n0 = blockIdx.x * BN;
  const int m0 = blockIdx.y * BM;

  const int r8 = tid >> 3;            // 0..31   (B tile fp32 loads)
  const int c4 = (tid & 7) << 2;
  const int r4 = tid >> 2;            // 0..63   (A tile bf16 16B copies)
  const int q8 = (tid & 3) << 3;      // 0,8,16,24 halfs

  v8f acc[2][4] = {};
  float4 bReg[4];
#if !HAVE_ASYNC_LDS
  uint4 aReg[2];
#endif

  // prologue: tile 0
#if HAVE_ASYNC_LDS
#pragma unroll
  for (int s = 0; s < 2; ++s)
    async_copy16(Z + (size_t)(m0 + r4 + s * 64) * CC + q8, &As[0][(r4 + s * 64) * LDA + q8]);
#else
#pragma unroll
  for (int s = 0; s < 2; ++s)
    aReg[s] = *reinterpret_cast<const uint4*>(Z + (size_t)(m0 + r4 + s * 64) * CC + q8);
#endif
#pragma unroll
  for (int s = 0; s < 4; ++s)
    bReg[s] = *reinterpret_cast<const float4*>(Wo + (size_t)(n0 + r8 + s * 32) * CC + c4);
#pragma unroll
  for (int s = 0; s < 4; ++s)
    cvt_store8(&Bs[0][(r8 + s * 32) * LDA + c4], bReg[s]);
#if HAVE_ASYNC_LDS
  __builtin_amdgcn_s_wait_asynccnt(0);
#else
#pragma unroll
  for (int s = 0; s < 2; ++s)
    *reinterpret_cast<uint4*>(&As[0][(r4 + s * 64) * LDA + q8]) = aReg[s];
#endif
  __syncthreads();

  for (int it = 0; it < NKIT; ++it) {
    const int cur = it & 1;
    const int kt  = (it + 1) * BK;
    if (it + 1 < NKIT) {
#if HAVE_ASYNC_LDS
#pragma unroll
      for (int s = 0; s < 2; ++s)
        async_copy16(Z + (size_t)(m0 + r4 + s * 64) * CC + kt + q8,
                     &As[cur ^ 1][(r4 + s * 64) * LDA + q8]);
#else
#pragma unroll
      for (int s = 0; s < 2; ++s)
        aReg[s] = *reinterpret_cast<const uint4*>(Z + (size_t)(m0 + r4 + s * 64) * CC + kt + q8);
#endif
#pragma unroll
      for (int s = 0; s < 4; ++s)
        bReg[s] = *reinterpret_cast<const float4*>(Wo + (size_t)(n0 + r8 + s * 32) * CC + kt + c4);
    }
    mma_step(As[cur], Bs[cur], wm, wn, lane, acc);
    if (it + 1 < NKIT) {
#pragma unroll
      for (int s = 0; s < 4; ++s)
        cvt_store8(&Bs[cur ^ 1][(r8 + s * 32) * LDA + c4], bReg[s]);
#if !HAVE_ASYNC_LDS
#pragma unroll
      for (int s = 0; s < 2; ++s)
        *reinterpret_cast<uint4*>(&As[cur ^ 1][(r4 + s * 64) * LDA + q8]) = aReg[s];
#endif
    }
#if HAVE_ASYNC_LDS
    __builtin_amdgcn_s_wait_asynccnt(0);   // wave's async copies for next tile done
#endif
    __syncthreads();
  }

  const int lane15 = lane & 15;
  const int hi8 = (lane >> 4) * 8;
#pragma unroll
  for (int i = 0; i < 2; ++i) {
#pragma unroll
    for (int j = 0; j < 4; ++j) {
      const int n = n0 + wn * 64 + j * 16 + lane15;
      const size_t mb = (size_t)m0 + wm * 32 + i * 16 + hi8;
      const v8f a = acc[i][j];
#pragma unroll
      for (int r = 0; r < 8; ++r) Out[(mb + r) * CC + n] = a[r];
    }
  }
}

// ---------------------------------------------------------------------------
// WKV chunked scan. State (aa, bb, pp) represents A = aa*e^pp, B = bb*e^pp,
// recurrence A' = e^w * A + e^k * v (per channel decay w = spatial_decay[c]/T).
// All exp arguments are <= 0 (max-normalized) -> __expf (v_exp_f32) is safe.
// ---------------------------------------------------------------------------

// Pass 1: per-chunk summaries with zero initial state. thread per (chunk,b,c).
__global__ __launch_bounds__(256)
void wkv_pass1(const float* __restrict__ K, const float* __restrict__ V,
               const float* __restrict__ sdecay,
               float* __restrict__ sa, float* __restrict__ sb, float* __restrict__ sp) {
  const int idx = blockIdx.x * 256 + threadIdx.x;       // NCH*BB*CC
  const int c  = idx % CC;
  const int rb = idx / CC;
  const int b  = rb % BB;
  const int ch = rb / BB;
  const float w = sdecay[c] * (1.0f / TT);

  float aa = 0.0f, bb = 0.0f, pp = -1e38f;
  size_t base = ((size_t)b * TT + (size_t)ch * CHL) * CC + c;
  for (int i = 0; i < CHL; ++i) {
    const float kt = K[base];
    const float vt = V[base];
    base += CC;
    const float ww2 = w + pp;
    const float p2  = fmaxf(ww2, kt);
    const float e1  = __expf(ww2 - p2);
    const float e2  = __expf(kt - p2);
    aa = e1 * aa + e2 * vt;
    bb = e1 * bb + e2;
    pp = p2;
  }
  const int o = (ch * BB + b) * CC + c;
  sa[o] = aa; sb[o] = bb; sp[o] = pp;
}

// Pass 2: in-place exclusive prefix combine over chunks; thread per (b,c).
__global__ __launch_bounds__(256)
void wkv_pass2(const float* __restrict__ sdecay,
               float* __restrict__ sa, float* __restrict__ sb, float* __restrict__ sp) {
  const int idx = blockIdx.x * 256 + threadIdx.x;       // BB*CC
  const int c = idx % CC;
  const int b = idx / CC;
  const float decay = sdecay[c] * (1.0f / TT) * (float)CHL;

  float aa = 0.0f, bb = 0.0f, pp = -1e38f;
  for (int ch = 0; ch < NCH; ++ch) {
    const int o = (ch * BB + b) * CC + c;
    const float ca = sa[o], cb = sb[o], cp = sp[o];
    sa[o] = aa; sb[o] = bb; sp[o] = pp;                 // exclusive prefix
    const float p1 = pp + decay;
    const float p  = fmaxf(p1, cp);
    const float e1 = __expf(p1 - p);
    const float e2 = __expf(cp - p);
    aa = e1 * aa + e2 * ca;
    bb = e1 * bb + e2 * cb;
    pp = p;
  }
}

// Pass 3: replay chunk with correct incoming state, emit z = sigmoid(r)*y (bf16).
__global__ __launch_bounds__(256)
void wkv_pass3(const float* __restrict__ K, const float* __restrict__ V,
               const unsigned short* __restrict__ SR,
               const float* __restrict__ sdecay, const float* __restrict__ sfirst,
               const float* __restrict__ sa, const float* __restrict__ sb,
               const float* __restrict__ sp,
               unsigned short* __restrict__ Zb) {
  const int idx = blockIdx.x * 256 + threadIdx.x;       // NCH*BB*CC
  const int c  = idx % CC;
  const int rb = idx / CC;
  const int b  = rb % BB;
  const int ch = rb / BB;
  const float w = sdecay[c] * (1.0f / TT);
  const float u = sfirst[c] * (1.0f / TT);

  const int o = (ch * BB + b) * CC + c;
  float aa = sa[o], bb = sb[o], pp = sp[o];

  size_t base = ((size_t)b * TT + (size_t)ch * CHL) * CC + c;
  for (int i = 0; i < CHL; ++i) {
    const float kt = K[base];
    const float vt = V[base];
    // output using incoming state
    const float ww = u + kt;
    const float p  = fmaxf(pp, ww);
    const float e1 = __expf(pp - p);
    const float e2 = __expf(ww - p);
    const float y  = (e1 * aa + e2 * vt) / (e1 * bb + e2);
    const float sr = bf2f(SR[base]);
    Zb[base] = f2bf(sr * y);
    // state update
    const float ww2 = w + pp;
    const float p2  = fmaxf(ww2, kt);
    const float e1b = __expf(ww2 - p2);
    const float e2b = __expf(kt - p2);
    aa = e1b * aa + e2b * vt;
    bb = e1b * bb + e2b;
    pp = p2;
    base += CC;
  }
}

// ---------------------------------------------------------------------------
// Host launcher
// ---------------------------------------------------------------------------
extern "C" void kernel_launch(void* const* d_in, const int* in_sizes, int n_in,
                              void* d_out, int out_size, void* d_ws, size_t ws_size,
                              hipStream_t stream) {
  (void)in_sizes; (void)n_in; (void)out_size; (void)ws_size;
  const float* x   = (const float*)d_in[0];
  const float* wk  = (const float*)d_in[1];
  const float* wv  = (const float*)d_in[2];
  const float* wr  = (const float*)d_in[3];
  const float* wo  = (const float*)d_in[4];
  const float* sd  = (const float*)d_in[5];   // spatial_decay
  const float* sf  = (const float*)d_in[6];   // spatial_first
  float* out = (float*)d_out;

  // Workspace layout
  char* ws = (char*)d_ws;
  const size_t elems = (size_t)MM * CC;            // 25,165,824
  float* kf  = (float*)ws;                          ws += elems * sizeof(float);   // 96 MB
  float* vf  = (float*)ws;                          ws += elems * sizeof(float);   // 96 MB
  unsigned short* srb = (unsigned short*)ws;        ws += elems * sizeof(unsigned short); // 48 MB
  unsigned short* zb  = (unsigned short*)ws;        ws += elems * sizeof(unsigned short); // 48 MB
  const size_t nsum = (size_t)NCH * BB * CC;       // 393,216
  float* sa = (float*)ws;                           ws += nsum * sizeof(float);
  float* sb = (float*)ws;                           ws += nsum * sizeof(float);
  float* sp = (float*)ws;                           ws += nsum * sizeof(float);

  // 1) k, v, sigmoid(r)
  {
    dim3 grid(NCAT / BN, MM / BM);                 // (18, 256)
    hipLaunchKernelGGL(vrwkv_gemm_kvr, grid, dim3(256), 0, stream,
                       x, wk, wv, wr, kf, vf, srb);
  }
  // 2) chunked WKV scan
  {
    const int n1 = NCH * BB * CC;                  // 393,216
    hipLaunchKernelGGL(wkv_pass1, dim3(n1 / 256), dim3(256), 0, stream,
                       kf, vf, sd, sa, sb, sp);
    const int n2 = BB * CC;                        // 6,144
    hipLaunchKernelGGL(wkv_pass2, dim3(n2 / 256), dim3(256), 0, stream,
                       sd, sa, sb, sp);
    hipLaunchKernelGGL(wkv_pass3, dim3(n1 / 256), dim3(256), 0, stream,
                       kf, vf, srb, sd, sf, sa, sb, sp, zb);
  }
  // 3) out = z @ Wo^T
  {
    dim3 grid(CC / BN, MM / BM);                   // (6, 256)
    hipLaunchKernelGGL(vrwkv_gemm_out, grid, dim3(256), 0, stream,
                       zb, wo, out);
  }
}
